// PCup_62792421868383
// MI455X (gfx1250) — compile-verified, matrix operand order checked by reading, not run on hardware
//
#include <hip/hip_runtime.h>

// ---------------- problem constants ----------------
#define NB       16
#define NPTS     16384
#define NCOARSE  1024
#define LAT      256
#define NDENSE   16384
#define CINF     261            // 3 (fg) + 2 (seed) + 256 (latent)
#define KPADF    288            // CINF padded to multiple of 32
#define FPS_SPLIT 16
#define FPS_SLICE (NPTS / FPS_SPLIT)   // 1024
#define FPS_PPT   (FPS_SLICE / 256)    // 4 points per thread

// ---------------- workspace layout (bytes) ----------------
#define WS_IDX    4096          // int[B*NCOARSE]
#define WS_WB     73728         // bf16 padded weights, 16B aligned
#define WS_W1P    (WS_WB)                 //  64x32   =  4096 B
#define WS_W2P    (WS_WB + 4096)          // 128x64   = 16384 B
#define WS_W3P    (WS_WB + 20480)         // 256x128  = 65536 B
#define ENC_WBYTES 86016                  // W1p+W2p+W3p blob
#define WS_F1P    (WS_WB + 86016)         //  64x288  = 36864 B
#define WS_F2P    (WS_WB + 122880)        // 128x64   = 16384 B
#define WS_F3P    (WS_WB + 139264)        // 256x128  = 65536 B
#define FOLD_WBYTES 81920                 // F2p|F3p blob
#define WS_BN     278528        // fused BN scale/bias (f32), 8 arrays
#define WS_BFEAT  393216

typedef __attribute__((ext_vector_type(16))) __bf16 v16bf;
typedef __attribute__((ext_vector_type(8)))  float  v8f;
typedef unsigned int uint32x4 __attribute__((ext_vector_type(4)));
typedef int int32x8 __attribute__((ext_vector_type(8)));
typedef int int32x4 __attribute__((ext_vector_type(4)));

// ---------------- bf16 helpers (raw u16 storage) ----------------
__device__ __forceinline__ unsigned short f2bf(float f) {
  unsigned u = __float_as_uint(f);
  u += 0x7FFFu + ((u >> 16) & 1u);          // round-to-nearest-even
  return (unsigned short)(u >> 16);
}
__device__ __forceinline__ float bf2f(unsigned short h) {
  return __uint_as_float(((unsigned)h) << 16);
}

// ---------------- TDM: 1-D tile, 8B elements, global -> LDS ----------------
__device__ __forceinline__ void tdm_load_1d(const void* gsrc, void* lds_dst,
                                            unsigned bytes) {
#if __has_builtin(__builtin_amdgcn_tensor_load_to_lds)
  const unsigned units = bytes >> 3;                    // 8B units
  const unsigned long long ga = (unsigned long long)gsrc;
  const unsigned lo = (unsigned)(unsigned long long)lds_dst;  // LDS byte offset
  uint32x4 g0;
  g0[0] = 1u;                                           // count=1 (valid D#)
  g0[1] = lo;                                           // lds_addr
  g0[2] = (unsigned)(ga & 0xffffffffu);                 // global_addr[31:0]
  g0[3] = (unsigned)((ga >> 32) & 0x1ffffffu) | (2u << 30);  // [56:32]|type=2
  int32x8 g1;
  g1[0] = (int)(3u << 16);                              // data_size=8B, mask=0
  g1[1] = (int)((units & 0xffffu) << 16);               // tensor_dim0[15:0]
  g1[2] = (int)(((units >> 16) & 0xffffu) | (1u << 16));// dim0 hi | tensor_dim1=1
  g1[3] = (int)(units << 16);                           // tile_dim0
  g1[4] = 0;                                            // tile_dim1/2 unused
  g1[5] = (int)units;                                   // tensor_dim0_stride lo
  g1[6] = 0; g1[7] = 0;
  int32x4 g2 = {0, 0, 0, 0};
  int32x4 g3 = {0, 0, 0, 0};
#if __clang_major__ >= 23
  int32x8 g4 = {0, 0, 0, 0, 0, 0, 0, 0};
  __builtin_amdgcn_tensor_load_to_lds(g0, g1, g2, g3, g4, 0);
#else
  __builtin_amdgcn_tensor_load_to_lds(g0, g1, g2, g3, 0);
#endif
  __builtin_amdgcn_s_wait_tensorcnt(0);
#else
  (void)gsrc; (void)lds_dst; (void)bytes;
#endif
}

// weight staging: TDM if available, cooperative copy fallback
__device__ __forceinline__ void stage_weights(unsigned short* dst,
                                              const unsigned short* src,
                                              unsigned bytes, int tid, int nthr,
                                              int wid) {
#if __has_builtin(__builtin_amdgcn_tensor_load_to_lds)
  (void)tid; (void)nthr;
  if (wid == 0) tdm_load_1d(src, dst, bytes);
#else
  (void)wid;
  const uint4* s4 = (const uint4*)src;
  uint4* d4 = (uint4*)dst;
  for (unsigned i = tid; i < bytes / 16; i += nthr) d4[i] = s4[i];
#endif
  __syncthreads();
}

// ---------------- WMMA fragments (wave32, 16x16x32 bf16) ----------------
// 16-bit operand layout (ISA 7.12.2): lane m=idx&15, half=idx>>4;
// regs 0..3 hold K=kc+half*8+{0..7}, regs 4..7 hold K=kc+16+half*8+{0..7}.
// Both chunks are 8 consecutive bf16 -> two aligned 16B loads.
__device__ __forceinline__ v16bf frag_ld(const unsigned short* r) {
  struct Q { uint4 a, b; } q;
  q.a = *(const uint4*)(r);
  q.b = *(const uint4*)(r + 16);
  return __builtin_bit_cast(v16bf, q);
}

// one 16(pts) x 16(channels) tile; act LDS [16][sAct], wgt [O][sW] padded bf16
__device__ __forceinline__ v8f wmma_tile(const unsigned short* act, int sAct,
                                         const unsigned short* wgt, int sW,
                                         int Kpad, int colg, int lane) {
  const int half8 = ((lane >> 4) & 1) * 8;
  const unsigned short* ar = act + (lane & 15) * sAct + half8;
  const unsigned short* br = wgt + (size_t)colg * sW + half8;
  v8f acc = {};
  for (int kc = 0; kc < Kpad; kc += 32) {
    v16bf a = frag_ld(ar + kc);
    v16bf b = frag_ld(br + kc);
    acc = __builtin_amdgcn_wmma_f32_16x16x32_bf16(false, a, false, b, (short)0,
                                                  acc, false, false);
  }
  return acc;
}

// fused epilogue: val = acc*A[col] + Bc[col], ReLU, bf16 -> LDS activation.
// C layout: col = tile*16 + (lane&15); rows = vgpr + ((lane<16)?0:8).
__device__ __forceinline__ void mlp_layer(const unsigned short* actIn, int sIn,
                                          int Kpad, const unsigned short* wgt,
                                          const float* __restrict__ A,
                                          const float* __restrict__ Bc, int O,
                                          unsigned short* actOut, int sOut,
                                          int lane) {
  const int rbase = (lane < 16) ? 0 : 8;
  for (int t = 0; t < O / 16; ++t) {
    const int colg = t * 16 + (lane & 15);
    v8f acc = wmma_tile(actIn, sIn, wgt, Kpad, Kpad, colg, lane);
    const float a = A[colg], bc = Bc[colg];
#pragma unroll
    for (int r = 0; r < 8; ++r) {
      float val = fmaf(acc[r], a, bc);
      actOut[(rbase + r) * sOut + colg] = f2bf(fmaxf(val, 0.f));
    }
  }
}

// ---------------- kernel 0a: pad+convert one weight matrix to bf16 ----------------
__global__ __launch_bounds__(256) void k_cvt(const float* __restrict__ src,
                                             unsigned short* __restrict__ dst,
                                             int O, int K, int Kpad) {
  int i = blockIdx.x * 256 + threadIdx.x;
  if (i >= O * Kpad) return;
  int o = i / Kpad, k = i - o * Kpad;
  dst[i] = (k < K) ? f2bf(src[o * K + k]) : (unsigned short)0;
}

// ---------------- kernel 0b: fold conv-bias + eval BN into (A, Bc) ----------------
// y = relu((x + bias)*sc + (be - m*sc)) with sc = g/sqrt(v+eps)
//   = relu(x*A + Bc),  A = sc,  Bc = bias*sc + be - m*sc
__global__ __launch_bounds__(256) void k_bn(const float* __restrict__ bias,
                                            const float* __restrict__ g,
                                            const float* __restrict__ be,
                                            const float* __restrict__ m,
                                            const float* __restrict__ v,
                                            float* __restrict__ A,
                                            float* __restrict__ Bc, int C) {
  int i = blockIdx.x * 256 + threadIdx.x;
  if (i >= C) return;
  float sc = g[i] * rsqrtf(v[i] + 1e-5f);
  A[i] = sc;
  Bc[i] = fmaf(bias[i], sc, be[i] - m[i] * sc);
}

// ---------------- kernel 1: feat_global = max over N of raw coarse ----------------
__global__ __launch_bounds__(256) void k_max(const float* __restrict__ coarse,
                                             float* __restrict__ fg) {
  __shared__ float mx[3][256];
  const int b = blockIdx.x, t = threadIdx.x;
  float m0 = -3.4e38f, m1 = m0, m2 = m0;
  for (int n = t; n < NPTS; n += 256) {
    const float* p = coarse + ((size_t)b * NPTS + n) * 3;
    m0 = fmaxf(m0, p[0]); m1 = fmaxf(m1, p[1]); m2 = fmaxf(m2, p[2]);
  }
  mx[0][t] = m0; mx[1][t] = m1; mx[2][t] = m2;
  __syncthreads();
  for (int s = 128; s > 0; s >>= 1) {
    if (t < s) {
      mx[0][t] = fmaxf(mx[0][t], mx[0][t + s]);
      mx[1][t] = fmaxf(mx[1][t], mx[1][t + s]);
      mx[2][t] = fmaxf(mx[2][t], mx[2][t + s]);
    }
    __syncthreads();
  }
  if (t == 0) {
    fg[b * 3 + 0] = mx[0][0]; fg[b * 3 + 1] = mx[1][0]; fg[b * 3 + 2] = mx[2][0];
  }
}

// ---------------- kernel 2: encoder MLP 3->64->128->256 (WMMA) ----------------
#define EWAVES 8
__global__ __launch_bounds__(256) void k_encoder(
    const float* __restrict__ coarse, const unsigned short* __restrict__ wblob,
    const float* __restrict__ A1, const float* __restrict__ Bc1,
    const float* __restrict__ A2, const float* __restrict__ Bc2,
    const float* __restrict__ b3, float* __restrict__ cfeat,
    unsigned short* __restrict__ bfeat) {
  __shared__ __align__(16) unsigned short sW[ENC_WBYTES / 2];  // W1p|W2p|W3p
  __shared__ __align__(16) unsigned short sIN[EWAVES][16][32];
  __shared__ __align__(16) unsigned short sA1[EWAVES][16][64];
  __shared__ __align__(16) unsigned short sA2[EWAVES][16][128];
  const int wid = threadIdx.x >> 5, lane = threadIdx.x & 31;
  const int p0 = (blockIdx.x * EWAVES + wid) * 16;     // 16 points per wave
  stage_weights(sW, wblob, ENC_WBYTES, threadIdx.x, 256, wid);
  const unsigned short* sW1 = sW;                      //  64x32
  const unsigned short* sW2 = sW + 2048;               // 128x64
  const unsigned short* sW3 = sW + 10240;              // 256x128
  if (lane < 16) {
    const float* p = coarse + (size_t)(p0 + lane) * 3;
    unsigned short* r = sIN[wid][lane];
    r[0] = f2bf(p[0]); r[1] = f2bf(p[1]); r[2] = f2bf(p[2]);
#pragma unroll
    for (int c = 3; c < 32; ++c) r[c] = 0;
  }
  __syncthreads();
  mlp_layer(&sIN[wid][0][0], 32, 32, sW1, A1, Bc1, 64, &sA1[wid][0][0], 64, lane);
  __syncthreads();
  mlp_layer(&sA1[wid][0][0], 64, 64, sW2, A2, Bc2, 128, &sA2[wid][0][0], 128, lane);
  __syncthreads();
  const int rbase = (lane < 16) ? 0 : 8;
  for (int t = 0; t < 16; ++t) {                       // 256 out channels
    const int colg = t * 16 + (lane & 15);
    v8f acc = wmma_tile(&sA2[wid][0][0], 128, sW3, 128, 128, colg, lane);
    const float bb = b3[colg];
#pragma unroll
    for (int r = 0; r < 8; ++r) {
      float val = acc[r] + bb;
      size_t o = (size_t)(p0 + rbase + r) * LAT + colg;
      cfeat[o] = val;
      if (bfeat) bfeat[o] = f2bf(val);                 // L2-resident FPS copy
    }
  }
}

// ---------------- kernel 3: farthest point sampling ----------------
__device__ __forceinline__ float ldf(const float* p) { return *p; }
__device__ __forceinline__ float ldf(const unsigned short* p) { return bf2f(*p); }

__device__ __forceinline__ void load8(const unsigned short* p, float* o) {
  const uint4 q = *(const uint4*)p;                    // 8 bf16, 16B aligned
  o[0] = bf2f((unsigned short)q.x); o[1] = __uint_as_float(q.x & 0xffff0000u);
  o[2] = bf2f((unsigned short)q.y); o[3] = __uint_as_float(q.y & 0xffff0000u);
  o[4] = bf2f((unsigned short)q.z); o[5] = __uint_as_float(q.z & 0xffff0000u);
  o[6] = bf2f((unsigned short)q.w); o[7] = __uint_as_float(q.w & 0xffff0000u);
}
__device__ __forceinline__ void load8(const float* p, float* o) {
  const float4 a = *(const float4*)p, b = *(const float4*)(p + 4);
  o[0] = a.x; o[1] = a.y; o[2] = a.z; o[3] = a.w;
  o[4] = b.x; o[5] = b.y; o[6] = b.z; o[7] = b.w;
}

template <typename T>
__global__ __launch_bounds__(256) void k_fps(
    const T* __restrict__ feats, int* __restrict__ idx_out,
    float* __restrict__ bestd, int* __restrict__ besti,
    unsigned* __restrict__ counters, unsigned* __restrict__ flags,
    int* __restrict__ chosen) {
  __shared__ float pfeat[LAT];
  __shared__ float rd[256];
  __shared__ int ri[256];
  __shared__ int sh_cur;
  const int t = threadIdx.x;
  const int b = blockIdx.x / FPS_SPLIT;
  const int sl = blockIdx.x % FPS_SPLIT;
  const size_t base = (size_t)b * NPTS;
  if (sl == 0 && t == 0) idx_out[b * NCOARSE] = 0;     // deterministic start @0
  const T* rowp[FPS_PPT];
  float dmin[FPS_PPT];
#pragma unroll
  for (int j = 0; j < FPS_PPT; ++j) {
    rowp[j] = feats + (base + sl * FPS_SLICE + j * 256 + t) * LAT;
    dmin[j] = 1e10f;
  }
  int cur = 0;
  for (int it = 0; it < NCOARSE - 1; ++it) {
    pfeat[t] = ldf(feats + (base + cur) * LAT + t);    // stage query feature
    __syncthreads();
    float d[FPS_PPT];
#pragma unroll
    for (int j = 0; j < FPS_PPT; ++j) d[j] = 0.f;
    for (int c = 0; c < LAT; c += 8) {
      float pv[8];
#pragma unroll
      for (int u = 0; u < 8; ++u) pv[u] = pfeat[c + u];
#pragma unroll
      for (int j = 0; j < FPS_PPT; ++j) {
        float xv[8];
        load8(rowp[j] + c, xv);
#pragma unroll
        for (int u = 0; u < 8; ++u) {
          float x = xv[u] - pv[u];
          d[j] = fmaf(x, x, d[j]);
        }
      }
    }
    float bd = -1.f; int bi = 0;
#pragma unroll
    for (int j = 0; j < FPS_PPT; ++j) {
      dmin[j] = fminf(dmin[j], d[j]);
      if (dmin[j] > bd) { bd = dmin[j]; bi = sl * FPS_SLICE + j * 256 + t; }
    }
    rd[t] = bd; ri[t] = bi;
    __syncthreads();
    for (int s = 128; s > 0; s >>= 1) {                // first-index on ties
      if (t < s) {
        float d2 = rd[t + s]; int i2 = ri[t + s];
        if (d2 > rd[t] || (d2 == rd[t] && i2 < ri[t])) { rd[t] = d2; ri[t] = i2; }
      }
      __syncthreads();
    }
    if (t == 0) {
      bestd[b * FPS_SPLIT + sl] = rd[0];
      besti[b * FPS_SPLIT + sl] = ri[0];
      __threadfence();
      unsigned old = atomicAdd(&counters[b], 1u);      // monotonic split barrier
      if (old == (unsigned)(FPS_SPLIT * (it + 1) - 1)) {  // last arriver reduces
        float gd = -1.f; int gi = 0x7fffffff;
        for (int s2 = 0; s2 < FPS_SPLIT; ++s2) {
          float d2 = bestd[b * FPS_SPLIT + s2]; int i2 = besti[b * FPS_SPLIT + s2];
          if (d2 > gd || (d2 == gd && i2 < gi)) { gd = d2; gi = i2; }
        }
        chosen[b] = gi;
        idx_out[b * NCOARSE + it + 1] = gi;
        __threadfence();
        __hip_atomic_store(&flags[b], (unsigned)(it + 1), __ATOMIC_RELEASE,
                           __HIP_MEMORY_SCOPE_AGENT);
      }
      while (__hip_atomic_load(&flags[b], __ATOMIC_ACQUIRE,
                               __HIP_MEMORY_SCOPE_AGENT) < (unsigned)(it + 1)) {
        __builtin_amdgcn_s_sleep(2);
      }
      __threadfence();
      sh_cur = chosen[b];
    }
    __syncthreads();
    cur = sh_cur;
  }
}

// ---------------- kernel 4: folding MLP 261->64->128->256 + residual ----------------
#define FWAVES 4
__global__ __launch_bounds__(128) void k_fold(
    const float* __restrict__ cfeat, const int* __restrict__ idx,
    const float* __restrict__ fglob, const unsigned short* __restrict__ f1p,
    const unsigned short* __restrict__ fwblob,       // F2p|F3p
    const float* __restrict__ A1, const float* __restrict__ Bc1,
    const float* __restrict__ A2, const float* __restrict__ Bc2,
    const float* __restrict__ fb3, float* __restrict__ outp) {
  __shared__ __align__(16) unsigned short sW[FOLD_WBYTES / 2];   // F2p|F3p
  __shared__ __align__(16) unsigned short sIN[FWAVES][16][KPADF];
  __shared__ __align__(16) unsigned short sA1[FWAVES][16][64];
  __shared__ __align__(16) unsigned short sA2[FWAVES][16][128];
  const int wid = threadIdx.x >> 5, lane = threadIdx.x & 31;
  const int tile = blockIdx.x * FWAVES + wid;          // tile = b*NCOARSE + s
  const int b = tile >> 10;
  const int s = tile & (NCOARSE - 1);
  const int ip = idx[b * NCOARSE + s];
  const float* cfr = cfeat + ((size_t)b * NPTS + ip) * LAT;  // gathered latent
  stage_weights(sW, fwblob, FOLD_WBYTES, threadIdx.x, 128, wid);
  const unsigned short* sF2 = sW;                      // 128x64
  const unsigned short* sF3 = sW + 8192;               // 256x128
  // stage 16 rows: [fg(3), seed(2), latent(256), pad(27)] — latent shared
  for (int c = lane; c < LAT; c += 32) {
    unsigned short hv = f2bf(cfr[c]);
#pragma unroll
    for (int g = 0; g < 16; ++g) sIN[wid][g][5 + c] = hv;
  }
  if (lane < 16) {
    const int g = lane;
    unsigned short* r = sIN[wid][g];
    r[0] = f2bf(fglob[b * 3 + 0]);
    r[1] = f2bf(fglob[b * 3 + 1]);
    r[2] = f2bf(fglob[b * 3 + 2]);
    const float step = 0.1f / 3.0f;                    // linspace(-.05,.05,4)
    r[3] = f2bf(-0.05f + step * (float)(g & 3));
    r[4] = f2bf(-0.05f + step * (float)(g >> 2));
    for (int c = CINF; c < KPADF; ++c) r[c] = 0;
  }
  __syncthreads();
  mlp_layer(&sIN[wid][0][0], KPADF, KPADF, f1p, A1, Bc1, 64, &sA1[wid][0][0],
            64, lane);
  __syncthreads();
  mlp_layer(&sA1[wid][0][0], 64, 64, sF2, A2, Bc2, 128, &sA2[wid][0][0], 128,
            lane);
  __syncthreads();
  const int rbase = (lane < 16) ? 0 : 8;
  for (int t = 0; t < 16; ++t) {
    const int colg = t * 16 + (lane & 15);
    v8f acc = wmma_tile(&sA2[wid][0][0], 128, sF3, 128, 128, colg, lane);
    const float addv = fb3[colg] + cfr[colg];          // bias + residual latent
#pragma unroll
    for (int r = 0; r < 8; ++r) {
      size_t o = ((size_t)tile * 16 + rbase + r) * LAT + colg;
      outp[o] = acc[r] + addv;
    }
  }
}

// ---------------- host launcher ----------------
extern "C" void kernel_launch(void* const* d_in, const int* in_sizes, int n_in,
                              void* d_out, int out_size, void* d_ws,
                              size_t ws_size, hipStream_t stream) {
  (void)in_sizes; (void)n_in; (void)out_size;
  const float* coarse = (const float*)d_in[0];
  const float* W1 = (const float*)d_in[1];  const float* b1 = (const float*)d_in[2];
  const float* g1 = (const float*)d_in[3];  const float* be1 = (const float*)d_in[4];
  const float* m1 = (const float*)d_in[5];  const float* v1 = (const float*)d_in[6];
  const float* W2 = (const float*)d_in[7];  const float* b2 = (const float*)d_in[8];
  const float* g2 = (const float*)d_in[9];  const float* be2 = (const float*)d_in[10];
  const float* m2 = (const float*)d_in[11]; const float* v2 = (const float*)d_in[12];
  const float* W3 = (const float*)d_in[13]; const float* b3 = (const float*)d_in[14];
  const float* F1 = (const float*)d_in[15]; const float* fb1 = (const float*)d_in[16];
  const float* fg1 = (const float*)d_in[17]; const float* fbe1 = (const float*)d_in[18];
  const float* fm1 = (const float*)d_in[19]; const float* fv1 = (const float*)d_in[20];
  const float* F2 = (const float*)d_in[21]; const float* fb2 = (const float*)d_in[22];
  const float* fg2 = (const float*)d_in[23]; const float* fbe2 = (const float*)d_in[24];
  const float* fm2 = (const float*)d_in[25]; const float* fv2 = (const float*)d_in[26];
  const float* F3 = (const float*)d_in[27]; const float* fb3 = (const float*)d_in[28];

  float* finalOut = (float*)d_out;                       // [B, ND, 256]
  float* cfeat = finalOut + (size_t)NB * NDENSE * LAT;   // [B, N, 256]

  char* ws = (char*)d_ws;
  float* fgp         = (float*)ws;              // 48 f32
  unsigned* counters = (unsigned*)(ws + 256);   // [B]
  unsigned* flags    = (unsigned*)(ws + 512);   // [B]
  int* chosen        = (int*)(ws + 768);        // [B]
  float* bestd       = (float*)(ws + 1024);     // [B*SPLIT]
  int* besti         = (int*)(ws + 2048);       // [B*SPLIT]
  int* idxbuf        = (int*)(ws + WS_IDX);     // [B*NCOARSE]
  unsigned short* w1p = (unsigned short*)(ws + WS_W1P);
  unsigned short* w2p = (unsigned short*)(ws + WS_W2P);
  unsigned short* w3p = (unsigned short*)(ws + WS_W3P);
  unsigned short* f1p = (unsigned short*)(ws + WS_F1P);
  unsigned short* f2p = (unsigned short*)(ws + WS_F2P);
  unsigned short* f3p = (unsigned short*)(ws + WS_F3P);
  float* bn = (float*)(ws + WS_BN);   // eA1[64] eB1[64] eA2[128] eB2[128]
  float* eA1 = bn;        float* eB1 = bn + 64;
  float* eA2 = bn + 128;  float* eB2 = bn + 256;
  float* dA1 = bn + 384;  float* dB1 = bn + 448;
  float* dA2 = bn + 512;  float* dB2 = bn + 640;   // ends at +768 floats
  unsigned short* bfeat = (unsigned short*)(ws + WS_BFEAT);
  const size_t need = WS_BFEAT + (size_t)NB * NPTS * LAT * 2;  // ~134 MB (fits L2)
  const bool useBf = ws_size >= need;

  hipMemsetAsync(d_ws, 0, 4096, stream);  // re-arm FPS split barrier every call

  // pad+convert weights to bf16 images (tiny, L2-resident)
  k_cvt<<<(64 * 32 + 255) / 256, 256, 0, stream>>>(W1, w1p, 64, 3, 32);
  k_cvt<<<(128 * 64 + 255) / 256, 256, 0, stream>>>(W2, w2p, 128, 64, 64);
  k_cvt<<<(256 * 128 + 255) / 256, 256, 0, stream>>>(W3, w3p, 256, 128, 128);
  k_cvt<<<(64 * KPADF + 255) / 256, 256, 0, stream>>>(F1, f1p, 64, CINF, KPADF);
  k_cvt<<<(128 * 64 + 255) / 256, 256, 0, stream>>>(F2, f2p, 128, 64, 64);
  k_cvt<<<(256 * 128 + 255) / 256, 256, 0, stream>>>(F3, f3p, 256, 128, 128);

  // fold conv-bias + BN into per-channel (A, Bc)
  k_bn<<<1, 256, 0, stream>>>(b1, g1, be1, m1, v1, eA1, eB1, 64);
  k_bn<<<1, 256, 0, stream>>>(b2, g2, be2, m2, v2, eA2, eB2, 128);
  k_bn<<<1, 256, 0, stream>>>(fb1, fg1, fbe1, fm1, fv1, dA1, dB1, 64);
  k_bn<<<1, 256, 0, stream>>>(fb2, fg2, fbe2, fm2, fv2, dA2, dB2, 128);

  k_max<<<NB, 256, 0, stream>>>(coarse, fgp);

  k_encoder<<<(NB * NPTS) / (EWAVES * 16), 256, 0, stream>>>(
      coarse, w1p, eA1, eB1, eA2, eB2, b3, cfeat,
      useBf ? bfeat : (unsigned short*)nullptr);

  if (useBf)
    k_fps<unsigned short><<<NB * FPS_SPLIT, 256, 0, stream>>>(
        bfeat, idxbuf, bestd, besti, counters, flags, chosen);
  else
    k_fps<float><<<NB * FPS_SPLIT, 256, 0, stream>>>(
        cfeat, idxbuf, bestd, besti, counters, flags, chosen);

  k_fold<<<(NB * NCOARSE) / FWAVES, 128, 0, stream>>>(
      cfeat, idxbuf, fgp, f1p, f2p, dA1, dB1, dA2, dB2, fb3, finalOut);
}
